// CausalWanSelfAttention_87557203297229
// MI455X (gfx1250) — matrix operand order repaired and checked by reading, not compiled, not run on hardware
//
#include <hip/hip_runtime.h>
#include <math.h>

// ---------------------------------------------------------------------------
// Flash-attention with per-key-segment additive bias (log 0.1) for MI455X
// (gfx1250, wave32, WMMA). fp32 in/out, f16 WMMA math, fp32 online softmax
// in log2 domain (Q pre-scaled by 1/sqrt(D)*log2(e); v_exp_f32 is base-2).
//
//   S^T = K * Q^T          (v_wmma_f32_16x16x32_f16)
//   O^T += V^T * P^T       (v_wmma_f32_16x16x32_f16)
//
// P^T uses AMD's table key order, so the C-layout exp() results feed the
// next WMMA's B operand with zero cross-lane traffic, and the Q/K d slots are remapped so K LDS fragment
// loads are contiguous b128 pairs. Double-buffered LDS, one barrier per key
// block, global loads for the next block software-pipelined under compute.
// Assumes Lq % 128 == 0, D == 128; Lk generic (clamped rows + score mask).
// ---------------------------------------------------------------------------

typedef __attribute__((ext_vector_type(16))) _Float16 v16h;
typedef __attribute__((ext_vector_type(8)))  _Float16 h8;
typedef __attribute__((ext_vector_type(8)))  float    v8f;

#define LOG2_BIAS_F (-3.321928094887362f) /* log2(0.1) */

static __device__ __forceinline__ v8f wmma_f16(v16h a, v16h b, v8f c) {
  return __builtin_amdgcn_wmma_f32_16x16x32_f16(false, a, false, b,
                                                (short)0, c, false, false);
}

// pack two f32 -> one u32 of f16 halves (v_cvt_pk_rtz_f16_f32)
static __device__ __forceinline__ unsigned pk2(float lo, float hi) {
  return __builtin_bit_cast(unsigned, __builtin_amdgcn_cvt_pkrtz(lo, hi));
}

static __device__ __forceinline__ v16h cat8(h8 lo, h8 hi) {
  return __builtin_shufflevector(lo, hi, 0, 1, 2, 3, 4, 5, 6, 7,
                                 8, 9, 10, 11, 12, 13, 14, 15);
}

// lane-pair exchange (xor 16) via ds_swizzle immediate: xor=0x10, and=0x1f
static __device__ __forceinline__ float xor16f(float x) {
  return __builtin_bit_cast(
      float, __builtin_amdgcn_ds_swizzle(__builtin_bit_cast(int, x), 0x401f));
}

static __device__ __forceinline__ void load16(const float* p, float* t) {
#pragma unroll
  for (int u = 0; u < 4; ++u) {
    float4 x = ((const float4*)p)[u];
    t[4 * u + 0] = x.x; t[4 * u + 1] = x.y;
    t[4 * u + 2] = x.z; t[4 * u + 3] = x.w;
  }
}

__global__ __launch_bounds__(256)
void flash_bias_attn(const float* __restrict__ q, const float* __restrict__ k,
                     const float* __restrict__ v, const int* __restrict__ p_fe,
                     const int* __restrict__ p_bs, float* __restrict__ out,
                     int Lq, int Lk, int H) {
  constexpr int D    = 128;  // head dim
  constexpr int KB   = 32;   // key block
  constexpr int KS   = 136;  // klds row stride (halves), 16B-aligned rows
  constexpr int VS   = 40;   // vlds row stride (halves), 16B-aligned rows
  constexpr int KBUF = KB * KS;
  constexpr int VBUF = D * VS;

  __shared__ __align__(16) _Float16 klds[2 * KBUF]; // [key][d] f16
  __shared__ __align__(16) _Float16 vlds[2 * VBUF]; // [d][key] f16 (transposed)

  const int h    = blockIdx.y;
  const int lane = threadIdx.x & 31;
  const int wave = threadIdx.x >> 5;
  const int m    = lane & 15;   // tile row / q column
  const int hi   = lane >> 4;   // wave half
  const int qrow = blockIdx.x * 128 + wave * 16 + m;
  const float qscale = 0.08838834764831845f * 1.4426950408889634f;

  int fe = *p_fe;               // biased segment [fe, bs)
  int bs = *p_bs;
  fe = min(max(fe, 0), Lk);
  bs = min(max(bs, 0), Lk);

  // ---- Q tile -> f16 B-layout regs; slot map: lane hi loads d groups
  // {c*32+hi*16+0..15} so K A-fragments become contiguous.
  v16h qb[4];
  {
    const float* qp = q + ((size_t)qrow * H + h) * D;
#pragma unroll
    for (int c = 0; c < 4; ++c) {
      const float* p = qp + c * 32 + hi * 16;
      float t[16];
#pragma unroll
      for (int u = 0; u < 2; ++u) {
        float4 x = ((const float4*)p)[u];
        t[4 * u + 0] = x.x; t[4 * u + 1] = x.y;
        t[4 * u + 2] = x.z; t[4 * u + 3] = x.w;
        float4 y = ((const float4*)(p + 8))[u];
        t[8 + 4 * u + 0] = y.x; t[8 + 4 * u + 1] = y.y;
        t[8 + 4 * u + 2] = y.z; t[8 + 4 * u + 3] = y.w;
      }
      union { unsigned u[8]; v16h h; } qq;
#pragma unroll
      for (int j = 0; j < 8; ++j)
        qq.u[j] = pk2(t[2 * j] * qscale, t[2 * j + 1] * qscale);
      qb[c] = qq.h;
    }
  }

  float m_run = -INFINITY;
  float l_run = 0.0f;
  v8f acc[8]; // O^T: acc[dt] covers d = dt*16 + r + 8*hi, q column = m
#pragma unroll
  for (int i = 0; i < 8; ++i) {
    v8f z = {0.f, 0.f, 0.f, 0.f, 0.f, 0.f, 0.f, 0.f};
    acc[i] = z;
  }

  // staging assignment: 256 threads, 16 floats each of K and V
  const int srow = threadIdx.x >> 3;        // 0..31 key row
  const int sdg  = (threadIdx.x & 7) * 16;  // d group
  const int rot  = (threadIdx.x & 7) * 2;   // scatter stagger

  // ---- prologue: stage tile 0 into buffer 0 ----
  {
    const int crow  = min(srow, Lk - 1);
    const size_t gb = ((size_t)crow * H + h) * D + sdg;
    float kt[16], vt[16];
    load16(k + gb, kt);
    load16(v + gb, vt);
    uint4 w0, w1;
    w0.x = pk2(kt[0], kt[1]);   w0.y = pk2(kt[2], kt[3]);
    w0.z = pk2(kt[4], kt[5]);   w0.w = pk2(kt[6], kt[7]);
    w1.x = pk2(kt[8], kt[9]);   w1.y = pk2(kt[10], kt[11]);
    w1.z = pk2(kt[12], kt[13]); w1.w = pk2(kt[14], kt[15]);
    uint4* kd = (uint4*)&klds[srow * KS + sdg];
    kd[0] = w0; kd[1] = w1;
#pragma unroll
    for (int j = 0; j < 16; ++j) {
      const int jj = (j + rot) & 15;
      vlds[(sdg + jj) * VS + srow] = (_Float16)vt[jj];
    }
  }
  __syncthreads();

  for (int kb = 0; kb < Lk; kb += KB) {
    const int ib = (kb / KB) & 1;
    const _Float16* kl = klds + ib * KBUF;
    const _Float16* vl = vlds + ib * VBUF;
    const bool have_next = (kb + KB) < Lk;

    // ---- issue global loads for the NEXT tile (overlap with compute) ----
    float kt[16], vt[16];
    if (have_next) {
      const int crow  = min(kb + KB + srow, Lk - 1);
      const size_t gb = ((size_t)crow * H + h) * D + sdg;
      load16(k + gb, kt);
      load16(v + gb, vt);
      if (kb + 2 * KB < Lk) {
        const size_t gn =
            ((size_t)min(kb + 2 * KB + srow, Lk - 1) * H + h) * D + sdg;
        __builtin_prefetch(k + gn, 0, 3);
        __builtin_prefetch(v + gn, 0, 3);
      }
    }

    // ---------------- S^T = K * Q^T (two 16-key tiles) ----------------
    v8f st0 = {0.f, 0.f, 0.f, 0.f, 0.f, 0.f, 0.f, 0.f};
    v8f st1 = {0.f, 0.f, 0.f, 0.f, 0.f, 0.f, 0.f, 0.f};
    {
      const _Float16* kr0 = kl + m * KS + hi * 16;
      const _Float16* kr1 = kr0 + 16 * KS;
      v16h ak[8];
#pragma unroll
      for (int c = 0; c < 4; ++c) {
        ak[c]     = cat8(*(const h8*)(kr0 + c * 32),
                         *(const h8*)(kr0 + c * 32 + 8));
        ak[4 + c] = cat8(*(const h8*)(kr1 + c * 32),
                         *(const h8*)(kr1 + c * 32 + 8));
      }
#pragma unroll
      for (int c = 0; c < 4; ++c) {
        st0 = wmma_f16(ak[c], qb[c], st0);
        st1 = wmma_f16(ak[4 + c], qb[c], st1);
      }
    }

    // ---------------- bias + online softmax (log2 domain) ----------------
    float e0[8], e1[8];
    float bmax = -INFINITY;
    float bofs;
    const bool tail  = (kb + KB) > Lk;
    const bool strad = ((kb < fe) && (fe < kb + KB)) ||
                       ((kb < bs) && (bs < kb + KB));
    if (!tail && !strad) {
      bofs = ((kb >= fe) && (kb + KB <= bs)) ? LOG2_BIAS_F : 0.0f;
#pragma unroll
      for (int r = 0; r < 8; ++r) {
        e0[r] = st0[r];
        e1[r] = st1[r];
        bmax = fmaxf(bmax, fmaxf(e0[r], e1[r]));
      }
      bmax += bofs;
    } else {
      bofs = 0.0f;
#pragma unroll
      for (int r = 0; r < 8; ++r) {
        const int k0 = kb + r + hi * 8;
        const int k1 = k0 + 16;
        float s0 = st0[r] + ((k0 >= fe && k0 < bs) ? LOG2_BIAS_F : 0.0f);
        float s1 = st1[r] + ((k1 >= fe && k1 < bs) ? LOG2_BIAS_F : 0.0f);
        if (k0 >= Lk) s0 = -INFINITY;
        if (k1 >= Lk) s1 = -INFINITY;
        e0[r] = s0; e1[r] = s1;
        bmax = fmaxf(bmax, fmaxf(s0, s1));
      }
    }
    bmax = fmaxf(bmax, xor16f(bmax));
    const float mnew  = fmaxf(m_run, bmax);
    const float shift = mnew - bofs;
    float ssum = 0.0f;
#pragma unroll
    for (int r = 0; r < 8; ++r) {
      e0[r] = __builtin_amdgcn_exp2f(e0[r] - shift);
      e1[r] = __builtin_amdgcn_exp2f(e1[r] - shift);
      ssum += e0[r] + e1[r];
    }
    ssum += xor16f(ssum);
    if (__ballot(mnew > m_run) != 0ull) {   // wave-uniform: rescale rare
      const float alpha = __builtin_amdgcn_exp2f(m_run - mnew);
      l_run = l_run * alpha + ssum;
      m_run = mnew;
#pragma unroll
      for (int i = 0; i < 8; ++i) acc[i] *= alpha;
    } else {
      l_run += ssum;
    }

    // ---- P^T: C layout drops straight into B slots (table key order) --
    union { unsigned u[8]; v16h h; } pb;
#pragma unroll
    for (int t = 0; t < 4; ++t) {
      pb.u[t]     = pk2(e0[2 * t], e0[2 * t + 1]);
      pb.u[4 + t] = pk2(e1[2 * t], e1[2 * t + 1]);
    }

    // ---------------- O^T += V^T * P^T ----------------
    {
      v16h av[8];
#pragma unroll
      for (int dt = 0; dt < 8; ++dt) {
        const _Float16* vr = vl + (dt * 16 + m) * VS + hi * 8;
        av[dt] = cat8(*(const h8*)(vr), *(const h8*)(vr + 16));
      }
#pragma unroll
      for (int dt = 0; dt < 8; ++dt)
        acc[dt] = wmma_f16(av[dt], pb.h, acc[dt]);
    }

    // ---- store next tile into the other buffer ----
    if (have_next) {
      _Float16* kd = klds + (ib ^ 1) * KBUF;
      _Float16* vd = vlds + (ib ^ 1) * VBUF;
      uint4 w0, w1;
      w0.x = pk2(kt[0], kt[1]);   w0.y = pk2(kt[2], kt[3]);
      w0.z = pk2(kt[4], kt[5]);   w0.w = pk2(kt[6], kt[7]);
      w1.x = pk2(kt[8], kt[9]);   w1.y = pk2(kt[10], kt[11]);
      w1.z = pk2(kt[12], kt[13]); w1.w = pk2(kt[14], kt[15]);
      uint4* kq = (uint4*)&kd[srow * KS + sdg];
      kq[0] = w0; kq[1] = w1;
#pragma unroll
      for (int j = 0; j < 16; ++j) {
        const int jj = (j + rot) & 15;
        vd[(sdg + jj) * VS + srow] = (_Float16)vt[jj];
      }
    }
    __syncthreads();
  }

  // ---------------- write O[q][d] = acc / l ----------------
  const float invl = 1.0f / l_run;
  float* op = out + ((size_t)qrow * H + h) * D + hi * 8;
#pragma unroll
  for (int dt = 0; dt < 8; ++dt) {
    float4 a, b;
    a.x = acc[dt][0] * invl; a.y = acc[dt][1] * invl;
    a.z = acc[dt][2] * invl; a.w = acc[dt][3] * invl;
    b.x = acc[dt][4] * invl; b.y = acc[dt][5] * invl;
    b.z = acc[dt][6] * invl; b.w = acc[dt][7] * invl;
    *(float4*)(op + dt * 16)     = a;
    *(float4*)(op + dt * 16 + 4) = b;
  }
}

extern "C" void kernel_launch(void* const* d_in, const int* in_sizes, int n_in,
                              void* d_out, int out_size, void* d_ws,
                              size_t ws_size, hipStream_t stream) {
  (void)n_in; (void)out_size; (void)d_ws; (void)ws_size;
  const float* q  = (const float*)d_in[0];
  const float* k  = (const float*)d_in[1];
  const float* v  = (const float*)d_in[2];
  const int*   fe = (const int*)d_in[3];
  const int*   bs = (const int*)d_in[4];
  float*       o  = (float*)d_out;

  const int H = 12, D = 128;
  const int Lq = in_sizes[0] / (H * D); // B == 1
  const int Lk = in_sizes[1] / (H * D);

  dim3 grid(Lq / 128, H);
  dim3 block(256);
  flash_bias_attn<<<grid, block, 0, stream>>>(q, k, v, fe, bs, o, Lq, Lk, H);
}
